// KeyQueryAttention_10222022165085
// MI455X (gfx1250) — matrix operand order repaired and checked by reading, not compiled, not run on hardware
//
#include <hip/hip_runtime.h>
#include <hip/hip_bf16.h>

// ---------- types ----------
typedef __attribute__((ext_vector_type(16))) _Float16 v16h;
typedef __attribute__((ext_vector_type(8)))  _Float16 v8h;
typedef __attribute__((ext_vector_type(4)))  _Float16 v4h;
typedef __attribute__((ext_vector_type(8)))  float    v8f;

#define TOTAL_TOKENS 262144
#define DIM          512
#define LATENT       256
#define NUM_SEG      2048

#define TOKS    128       // tokens per block in logits kernel
#define KCHUNK  64        // K dims staged in LDS per pass (double buffered)
#define ROWH    72        // KCHUNK + 8 halves pad (stride 144B = 36 dwords -> conflict free)
#define NCHUNKS (DIM / KCHUNK)   // 8

__device__ __forceinline__ v8f zero8() {
    v8f z;
#pragma unroll
    for (int i = 0; i < 8; ++i) z[i] = 0.0f;
    return z;
}

// load 16 halves as two 8-half chunks at p and p+second_off (halves)
__device__ __forceinline__ v16h ld2x8(const _Float16* p, int second_off) {
    v8h lo = *(const v8h*)p;
    v8h hi = *(const v8h*)(p + second_off);
    v16h r;
#pragma unroll
    for (int i = 0; i < 8; ++i) { r[i] = lo[i]; r[i + 8] = hi[i]; }
    return r;
}

// monotonic float <-> uint order-preserving encoding (for atomicMax on floats)
__device__ __forceinline__ unsigned enc_f32(float f) {
    unsigned u = __float_as_uint(f);
    return (u & 0x80000000u) ? ~u : (u | 0x80000000u);
}
__device__ __forceinline__ float dec_f32(unsigned u) {
    u = (u & 0x80000000u) ? (u ^ 0x80000000u) : ~u;
    return __uint_as_float(u);
}

// ---------- staging helpers (global f32 -> regs -> f16 LDS) ----------
__device__ __forceinline__ void stage_regs(const float* __restrict__ src, float4 s[4]) {
#pragma unroll
    for (int i = 0; i < 4; ++i) s[i] = ((const float4*)src)[i];
}
__device__ __forceinline__ void stage_store(_Float16* dst, const float4 s[4]) {
#pragma unroll
    for (int i = 0; i < 4; ++i) {
        v4h h;
        h[0] = (_Float16)s[i].x; h[1] = (_Float16)s[i].y;
        h[2] = (_Float16)s[i].z; h[3] = (_Float16)s[i].w;
        ((v4h*)dst)[i] = h;
    }
}

// ---------- kernel 1: weights f32 -> f16, transposed to [N=256][K=512] ----------
__global__ __launch_bounds__(256) void prep_weights(const float* __restrict__ kw,
                                                    const float* __restrict__ qw,
                                                    _Float16* __restrict__ kT,
                                                    _Float16* __restrict__ qT) {
    int idx = blockIdx.x * 256 + threadIdx.x;        // 0 .. 2*131072-1
    int which = idx >> 17;
    int o = idx & 131071;                            // n*512 + k
    int n = o >> 9;
    int k = o & 511;
    const float* src = which ? qw : kw;
    _Float16* dst = which ? qT : kT;
    dst[o] = (_Float16)src[k * LATENT + n];
}

// ---------- kernel 2: fused dual-GEMM + per-token logit via WMMA ----------
//  512 threads = 16 waves; wave w owns N-tile w; 8 M-tiles x full K accumulated in regs.
//  seq is staged f32->f16 into double-buffered LDS, software-pipelined:
//    issue loads(ko+1) -> 32 WMMAs on buf[ko&1] -> cvt+store buf[(ko+1)&1] -> one barrier.
__global__ __launch_bounds__(512) void logits_kernel(const float* __restrict__ seq,
                                                     const _Float16* __restrict__ kT,
                                                     const _Float16* __restrict__ qT,
                                                     float* __restrict__ logits) {
    __shared__ _Float16 Ash[2][TOKS * ROWH];   // 2 x 18432 B
    __shared__ float    lds_logit[TOKS];

    const int tid  = threadIdx.x;           // 0..511
    const int wid  = tid >> 5;              // wave id == N-tile id (0..15)
    const int lane = tid & 31;
    const int l16  = lane & 15;
    const int lhi  = lane >> 4;             // 0 or 1
    const long tok0 = (long)blockIdx.x * TOKS;

    if (tid < TOKS) lds_logit[tid] = 0.0f;

    // per-wave accumulators: 8 M-tiles x {K,Q} (full K=512 accumulation)
    v8f acc_k[8], acc_q[8];
#pragma unroll
    for (int m = 0; m < 8; ++m) { acc_k[m] = zero8(); acc_q[m] = zero8(); }

    // B operand base: lane l16 holds column n = ntile*16+l16; contiguous 16 K per half-wave
    const _Float16* bk_base = kT + ((wid * 16 + l16) * DIM) + lhi * 16;
    const _Float16* bq_base = qT + ((wid * 16 + l16) * DIM) + lhi * 16;

    // staging map: thread -> (token row, 16-dim quarter) of the 128x64 chunk
    const float* gsrc = seq + (tok0 + (tid >> 2)) * DIM + (tid & 3) * 16;
    const int ldst_off = (tid >> 2) * ROWH + (tid & 3) * 16;
    // A operand base (halves offset within a buffer)
    const int a_off = l16 * ROWH + lhi * 16;

    // prologue: stage chunk 0
    float4 stg[4];
    stage_regs(gsrc, stg);
    stage_store(&Ash[0][ldst_off], stg);
    __syncthreads();

    for (int ko = 0; ko < NCHUNKS; ++ko) {
        const int cur = ko & 1;
        const bool more = (ko + 1) < NCHUNKS;
        if (more) stage_regs(gsrc + (ko + 1) * KCHUNK, stg);   // prefetch next chunk (in flight)

        const _Float16* abase = &Ash[cur][0] + a_off;
#pragma unroll
        for (int kt = 0; kt < KCHUNK / 32; ++kt) {             // 2 K-tiles per chunk
            int kg = ko * (KCHUNK / 32) + kt;
            v16h bk = ld2x8(bk_base + kg * 32, 8);
            v16h bq = ld2x8(bq_base + kg * 32, 8);
#pragma unroll
            for (int m = 0; m < 8; ++m) {
                v16h a = ld2x8(abase + m * 16 * ROWH + kt * 32, 8);
                acc_k[m] = __builtin_amdgcn_wmma_f32_16x16x32_f16(
                    false, a, false, bk, (short)0, acc_k[m], false, false);
                acc_q[m] = __builtin_amdgcn_wmma_f32_16x16x32_f16(
                    false, a, false, bq, (short)0, acc_q[m], false, false);
            }
        }

        if (more) stage_store(&Ash[1 - cur][ldst_off], stg);   // fill other buffer
        __syncthreads();                                       // single barrier per chunk
    }

    // logit[m-row] = sum_n Ktile * Qtile : elementwise product + lane reduce over N
#pragma unroll
    for (int m = 0; m < 8; ++m) {
#pragma unroll
        for (int c = 0; c < 8; ++c) {
            float v = acc_k[m][c] * acc_q[m][c];
            v += __shfl_xor(v, 1);
            v += __shfl_xor(v, 2);
            v += __shfl_xor(v, 4);
            v += __shfl_xor(v, 8);
            // lanes 0..15 hold row (m*16+c); lanes 16..31 hold row (m*16+8+c)
            if (l16 == 0)
                atomicAdd(&lds_logit[m * 16 + lhi * 8 + c], v);
        }
    }
    __syncthreads();
    if (tid < TOKS) logits[tok0 + tid] = lds_logit[tid];
}

// ---------- kernel 3: segment max (monotone-encoded uint atomicMax) ----------
__global__ __launch_bounds__(256) void segmax_kernel(const float* __restrict__ logits,
                                                     const int* __restrict__ seg,
                                                     unsigned* __restrict__ segmax) {
    int t = blockIdx.x * 256 + threadIdx.x;
    atomicMax(&segmax[seg[t]], enc_f32(logits[t]));
}

// ---------- kernel 4: segment softmax denominator ----------
__global__ __launch_bounds__(256) void denom_kernel(const float* __restrict__ logits,
                                                    const int* __restrict__ seg,
                                                    const unsigned* __restrict__ segmax,
                                                    float* __restrict__ denom) {
    int t = blockIdx.x * 256 + threadIdx.x;
    int s = seg[t];
    atomicAdd(&denom[s], __expf(logits[t] - dec_f32(segmax[s])));
}

// ---------- kernel 5: ragged weighted-sum pooling ----------
__global__ __launch_bounds__(256) void pool_kernel(const float* __restrict__ seq,
                                                   const float* __restrict__ bias,
                                                   const int* __restrict__ seg,
                                                   const float* __restrict__ logits,
                                                   const unsigned* __restrict__ segmax,
                                                   const float* __restrict__ denom,
                                                   float* __restrict__ out) {
    __shared__ float att_s[TOKS];
    __shared__ int   seg_s[TOKS];
    const int tid = threadIdx.x;
    const long tok0 = (long)blockIdx.x * TOKS;

    if (tid < TOKS) {
        int s = seg[tok0 + tid];
        seg_s[tid] = s;
        att_s[tid] = __expf(logits[tok0 + tid] - dec_f32(segmax[s])) / denom[s];
    }
    __syncthreads();

    const int d = tid * 2;                   // 256 threads x 2 dims = 512
    float2 b = *(const float2*)(bias + d);
    float ax = 0.0f, ay = 0.0f;
    int cur = seg_s[0];
    for (int t = 0; t < TOKS; ++t) {
        int s = seg_s[t];
        if (s != cur) {                      // sorted segments -> rare flush
            atomicAdd(&out[(long)cur * DIM + d],     ax);
            atomicAdd(&out[(long)cur * DIM + d + 1], ay);
            ax = 0.0f; ay = 0.0f; cur = s;
        }
        float2 x = *(const float2*)(seq + (tok0 + t) * DIM + d);
        float a = att_s[t];
        ax += (x.x + b.x) * a;
        ay += (x.y + b.y) * a;
    }
    atomicAdd(&out[(long)cur * DIM + d],     ax);
    atomicAdd(&out[(long)cur * DIM + d + 1], ay);
}

// ---------- launch ----------
extern "C" void kernel_launch(void* const* d_in, const int* in_sizes, int n_in,
                              void* d_out, int out_size, void* d_ws, size_t ws_size,
                              hipStream_t stream) {
    const float* seq     = (const float*)d_in[0];
    const float* key_w   = (const float*)d_in[1];
    const float* query_w = (const float*)d_in[2];
    const float* bias    = (const float*)d_in[3];
    const int*   segids  = (const int*)d_in[4];
    (void)n_in; (void)in_sizes; (void)ws_size;

    char* ws = (char*)d_ws;
    _Float16* kT     = (_Float16*)(ws);                 // 256*512*2 = 262144 B
    _Float16* qT     = (_Float16*)(ws + 262144);        // 262144 B
    float*    logits = (float*)(ws + 524288);           // 262144*4 = 1048576 B
    unsigned* segmax = (unsigned*)(ws + 1572864);       // 2048*4
    float*    denom  = (float*)(ws + 1581056);          // 2048*4

    hipMemsetAsync(ws + 1572864, 0, 16384, stream);                 // segmax(0 < enc(-inf)) + denom
    hipMemsetAsync(d_out, 0, (size_t)out_size * sizeof(float), stream);

    prep_weights<<<2 * 131072 / 256, 256, 0, stream>>>(key_w, query_w, kT, qT);
    logits_kernel<<<TOTAL_TOKENS / TOKS, 512, 0, stream>>>(seq, kT, qT, logits);
    segmax_kernel<<<TOTAL_TOKENS / 256, 256, 0, stream>>>(logits, segids, segmax);
    denom_kernel<<<TOTAL_TOKENS / 256, 256, 0, stream>>>(logits, segids, segmax, denom);
    pool_kernel<<<TOTAL_TOKENS / TOKS, 256, 0, stream>>>(seq, bias, segids, logits,
                                                         segmax, denom, (float*)d_out);
}
